// AttentionAgger_20435454395129
// MI455X (gfx1250) — matrix operand order repaired and checked by reading, not compiled, not run on hardware
//
#include <hip/hip_runtime.h>

// ---------------------------------------------------------------------------
// MI455X / gfx1250 projected-attention:
//   Qp = Q@WQ^T + bq ; Kp = K@WK^T + bk  (bf16 in workspace)
//   O  = softmax(mask ? NEG : Qp@Kp^T * 1/sqrt(512)) @ V   (flash, online softmax)
// Heavy math on v_wmma_f32_16x16x32_bf16 (wave32 WMMA), f32 accumulation.
// V/Q tiles staged via Tensor Data Mover (tensor_load_to_lds) with LDS double
// buffering + s_wait_tensorcnt pipelining; K tile staged manually transposed
// (feature-major) so B-operand lanes read adjacent krows (16-bit loads).
// Workspace: Qp (8MB) + Kp (8MB) + Vbf16 (8MB) = 24MB.
// ---------------------------------------------------------------------------

typedef __attribute__((ext_vector_type(16))) __bf16 bf16x16;
typedef __attribute__((ext_vector_type(8)))  float  f32x8;

#define NQR   8192
#define NKR   8192
#define DDIM  512
#define NEGV  (-4294967296.0f)   /* -(1<<32), matches reference NEG */

static __device__ inline unsigned short f2bfu(float f) {
  __bf16 h = (__bf16)f;
  return __builtin_bit_cast(unsigned short, h);
}
static __device__ inline unsigned pack2(float lo, float hi) {
  return (unsigned)f2bfu(lo) | ((unsigned)f2bfu(hi) << 16);
}

// ---------------------------------------------------------------------------
// Tensor Data Mover: 2D bf16 tile (rows x row_elems) global -> LDS, row-major.
// Descriptor bit layout per CDNA5 ISA §8.3/8.4 (D# group0/group1).
// This toolchain exposes the 6-arg builtin:
//   (uint32x4 g0, int32x8 g1, int32x4 g2, int32x4 g3, int32x8 extra, i32 cpol)
// ---------------------------------------------------------------------------
#if __has_builtin(__builtin_amdgcn_tensor_load_to_lds)
#define HAVE_TDM 1
typedef __attribute__((ext_vector_type(4))) unsigned u32x4;
typedef __attribute__((ext_vector_type(8))) int i32x8;
typedef __attribute__((ext_vector_type(4))) int i32x4;

static __device__ __attribute__((always_inline)) inline
void tdm_load_2d(const void* gaddr, unsigned lds_byte_off,
                 unsigned rows, unsigned row_elems) {
  const unsigned long long ga = (unsigned long long)gaddr;
  u32x4 g0;
  g0[0] = 1u;                                          // count=1, user desc
  g0[1] = lds_byte_off;                                // lds_addr [63:32]
  g0[2] = (unsigned)(ga & 0xFFFFFFFFu);                // global_addr lo
  g0[3] = (unsigned)((ga >> 32) & 0x1FFFFFFu)          // global_addr [56:32]
        | (2u << 30);                                  // type=2 ("image")
  i32x8 g1;
  g1[0] = (int)(1u << 16);                             // data_size=1 -> 2 bytes
  g1[1] = (int)((row_elems & 0xFFFFu) << 16);          // tensor_dim0 lo16
  g1[2] = (int)(((row_elems >> 16) & 0xFFFFu)          // tensor_dim0 hi16
        | ((rows & 0xFFFFu) << 16));                   // tensor_dim1 lo16
  g1[3] = (int)(((rows >> 16) & 0xFFFFu)               // tensor_dim1 hi16
        | ((row_elems & 0xFFFFu) << 16));              // tile_dim0
  g1[4] = (int)(rows & 0xFFFFu);                       // tile_dim1 (tile_dim2=0)
  g1[5] = (int)row_elems;                              // tensor_dim0_stride lo32
  g1[6] = 0;                                           // stride hi / dim1_stride
  g1[7] = 0;
  i32x4 gz4 = {0, 0, 0, 0};                            // groups 2/3 unused (2D)
  i32x8 gz8 = {0, 0, 0, 0, 0, 0, 0, 0};                // extra group (unused)
  __builtin_amdgcn_tensor_load_to_lds(g0, g1, gz4, gz4, gz8, 0);
}
static __device__ inline unsigned lds_off(const void* p) {
  // LDS aperture base has zero low 32 bits; addr[31:0] is the LDS byte offset.
  return (unsigned)(unsigned long long)p;
}
#else
#define HAVE_TDM 0
#endif

// ---------------------------------------------------------------------------
// Projection GEMM: Y[m,n] = sum_k X[m,k]*W[n,k] + b[n], Y stored bf16.
// grid = (M/16, 512/128), block = 256 (8 waves); 16 steps of wmma 16x16x32 bf16.
// ---------------------------------------------------------------------------
__global__ __launch_bounds__(256) void proj16(const float* __restrict__ X,
                                              const float* __restrict__ W,
                                              const float* __restrict__ bias,
                                              unsigned short* __restrict__ Y) {
  const int lane = threadIdx.x & 31;
  const int wave = threadIdx.x >> 5;
  const int half = lane >> 4;
  const int lm   = lane & 15;
  const int m0 = blockIdx.x * 16;
  const int n0 = blockIdx.y * 128 + wave * 16;

  f32x8 accA = {}, accB = {};
  const float* xrow = X + (size_t)(m0 + lm) * DDIM;

#pragma unroll
  for (int kt = 0; kt < 16; ++kt) {
    const int k0 = kt * 32;
    union { bf16x16 v; unsigned u[8]; } A;
    const int c0 = k0 + (half ? 8 : 0);
    const int c1 = k0 + (half ? 24 : 16);
#pragma unroll
    for (int i = 0; i < 4; ++i) {
      A.u[i]     = pack2(xrow[c0 + 2 * i], xrow[c0 + 2 * i + 1]);
      A.u[4 + i] = pack2(xrow[c1 + 2 * i], xrow[c1 + 2 * i + 1]);
    }
    union { bf16x16 v; unsigned u[8]; } B;
#pragma unroll
    for (int vv = 0; vv < 8; ++vv) {
      const int kk = (vv < 4) ? (c0 + 2 * vv) : (c1 + 2 * (vv - 4));
      float blo = W[(size_t)(n0 + lm) * DDIM + kk];
      float bhi = W[(size_t)(n0 + lm) * DDIM + kk + 1];
      B.u[vv] = pack2(blo, bhi);
    }
    if (kt & 1)
      accB = __builtin_amdgcn_wmma_f32_16x16x32_bf16(false, A.v, false, B.v,
                                                     (short)0, accB, false, false);
    else
      accA = __builtin_amdgcn_wmma_f32_16x16x32_bf16(false, A.v, false, B.v,
                                                     (short)0, accA, false, false);
  }
  f32x8 acc = accA + accB;

  const float bn = bias[n0 + lm];
#pragma unroll
  for (int r = 0; r < 8; ++r) {
    const int m = m0 + r + 8 * half;
    Y[(size_t)m * DDIM + n0 + lm] = f2bfu(acc[r] + bn);
  }
}

// ---------------------------------------------------------------------------
// V f32 -> bf16
// ---------------------------------------------------------------------------
__global__ __launch_bounds__(256) void cvt_bf16(const float* __restrict__ V,
                                                unsigned short* __restrict__ Vb) {
  const size_t i = ((size_t)blockIdx.x * blockDim.x + threadIdx.x) * 4;
  float4 f = *(const float4*)(V + i);
  uint2 o;
  o.x = pack2(f.x, f.y);
  o.y = pack2(f.z, f.w);
  *(uint2*)(Vb + i) = o;
}

// ---------------------------------------------------------------------------
// Flash attention. 256 threads (8 waves); each wave owns 16 Q rows and the
// full 512-wide output (32 f32 C tiles). K blocks of 32 rows (= WMMA depth).
// LDS: Q 128x512 bf16 (128K) | Kt [512][32] (32K) | V x2 (64K) | P (8K) = 232K.
// ---------------------------------------------------------------------------
#define SM_Q   0
#define SM_KT  (128 * DDIM * 2)                 // 131072
#define SM_V0  (SM_KT + DDIM * 32 * 2)          // 163840
#define SM_V1  (SM_V0 + 32 * DDIM * 2)          // 196608
#define SM_P   (SM_V1 + 32 * DDIM * 2)          // 229376
#define SM_SZ  (SM_P + 8 * 16 * 32 * 2)         // 237568 bytes

__global__ __launch_bounds__(256, 1) void flash_attn(const unsigned short* __restrict__ Qp,
                                                     const unsigned short* __restrict__ Kp,
                                                     const unsigned short* __restrict__ Vb,
                                                     const unsigned char* __restrict__ mask,
                                                     float* __restrict__ Out) {
  extern __shared__ char smem[];
  unsigned short* Qs  = (unsigned short*)(smem + SM_Q);   // [128][512] row-major
  unsigned short* Kt  = (unsigned short*)(smem + SM_KT);  // [512 feat][32 krow]
  unsigned short* Vs0 = (unsigned short*)(smem + SM_V0);  // [32][512] buf 0
  unsigned short* Vs1 = (unsigned short*)(smem + SM_V1);  // [32][512] buf 1
  unsigned short* Ps  = (unsigned short*)(smem + SM_P);   // [8][16][32]

  const int tid  = threadIdx.x;
  const int lane = tid & 31;
  const int wave = tid >> 5;
  const int half = lane >> 4;
  const int lm   = lane & 15;
  const int blockQ0 = blockIdx.x * 128;
  const int q0 = blockQ0 + wave * 16;

  // ---- prologue staging: Q block + first V tile
#if HAVE_TDM
  if (wave == 0) {
    tdm_load_2d(Qp + (size_t)blockQ0 * DDIM, lds_off(Qs), 128, DDIM);
    tdm_load_2d(Vb, lds_off(Vs0), 32, DDIM);
    __builtin_amdgcn_s_wait_tensorcnt(0);
  }
#else
  {
    const uint4* src = (const uint4*)(Qp + (size_t)blockQ0 * DDIM);
    uint4* dst = (uint4*)Qs;
#pragma unroll
    for (int i = 0; i < 32; ++i)
      dst[tid + i * 256] = src[tid + i * 256];
  }
#endif
  __syncthreads();

  float mrow[8], lrow[8];
#pragma unroll
  for (int r = 0; r < 8; ++r) { mrow[r] = -__builtin_inff(); lrow[r] = 0.0f; }

  f32x8 o[32];
#pragma unroll
  for (int t = 0; t < 32; ++t) o[t] = (f32x8){};

  const float scale = 0.044194173824159216f;     // 1/sqrt(512)

  for (int it = 0; it < NKR / 32; ++it) {
    const int kb = it * 32;
    // ---- stage Kp block transposed (feature-major) -- manual (TDM can't transpose)
    {
      const int n = tid >> 3;                    // krow 0..31
      const int kbase = (tid & 7) * 64;          // feature base
      const unsigned short* src = Kp + (size_t)(kb + n) * DDIM + kbase;
#pragma unroll
      for (int c = 0; c < 8; ++c) {
        uint4 d = *(const uint4*)(src + c * 8);
        const unsigned short* e = (const unsigned short*)&d;
#pragma unroll
        for (int j = 0; j < 8; ++j)
          Kt[(kbase + c * 8 + j) * 32 + n] = e[j];
      }
    }
#if HAVE_TDM
    // ---- V pipeline: V(it) already in flight/landed; kick V(it+1) into the
    // other buffer (its readers finished at the previous end-of-iter barrier).
    if (wave == 0) {
      __builtin_amdgcn_s_wait_tensorcnt(0);      // V(it) landed
      if (it + 1 < NKR / 32)
        tdm_load_2d(Vb + (size_t)(kb + 32) * DDIM,
                    ((it + 1) & 1) ? lds_off(Vs1) : lds_off(Vs0), 32, DDIM);
    }
    const unsigned short* Vs = (it & 1) ? Vs1 : Vs0;
#else
    {
      const uint4* vsrc = (const uint4*)(Vb + (size_t)kb * DDIM);
      uint4* vdst = (uint4*)Vs0;
#pragma unroll
      for (int c = 0; c < 8; ++c)
        vdst[tid + c * 256] = vsrc[tid + c * 256];
    }
    const unsigned short* Vs = Vs0;
#endif
    // prefetch next mask block (largest HBM stream)
    if (kb + 32 < NKR)
      __builtin_prefetch(&mask[(size_t)(q0 + lm) * NKR + kb + 32], 0, 1);
    __syncthreads();

    // ---- S = Qp(16x512) @ Kp^T(512x32): 4 independent WMMA chains to cover
    // the bf16 WMMA->WMMA RAW hazard window (5 slots) with real work.
    f32x8 sa0 = {}, sb0 = {}, sa1 = {}, sb1 = {};
#pragma unroll
    for (int kt = 0; kt < 16; ++kt) {
      union { bf16x16 v; uint4 u[2]; } A;
      const unsigned short* qrow = Qs + (wave * 16 + lm) * DDIM + kt * 32;
      A.u[0] = *(const uint4*)(qrow + (half ? 8 : 0));
      A.u[1] = *(const uint4*)(qrow + (half ? 24 : 16));
      union { bf16x16 v; unsigned u[8]; } B0, B1;
      const unsigned short* kcol = Kt + (kt * 32) * 32 + lm;
#pragma unroll
      for (int vv = 0; vv < 8; ++vv) {
        const int kk = (vv < 4) ? (8 * half + 2 * vv) : (16 + 8 * half + 2 * (vv - 4));
        B0.u[vv] = (unsigned)kcol[kk * 32] | ((unsigned)kcol[(kk + 1) * 32] << 16);
        B1.u[vv] = (unsigned)kcol[kk * 32 + 16] | ((unsigned)kcol[(kk + 1) * 32 + 16] << 16);
      }
      if (kt & 1) {
        sb0 = __builtin_amdgcn_wmma_f32_16x16x32_bf16(false, A.v, false, B0.v,
                                                      (short)0, sb0, false, false);
        sb1 = __builtin_amdgcn_wmma_f32_16x16x32_bf16(false, A.v, false, B1.v,
                                                      (short)0, sb1, false, false);
      } else {
        sa0 = __builtin_amdgcn_wmma_f32_16x16x32_bf16(false, A.v, false, B0.v,
                                                      (short)0, sa0, false, false);
        sa1 = __builtin_amdgcn_wmma_f32_16x16x32_bf16(false, A.v, false, B1.v,
                                                      (short)0, sa1, false, false);
      }
    }
    f32x8 s0 = sa0 + sb0;
    f32x8 s1 = sa1 + sb1;

    // ---- scale + mask (mask true -> NEG)
#pragma unroll
    for (int r = 0; r < 8; ++r) {
      s0[r] *= scale;
      s1[r] *= scale;
      const int m = q0 + r + 8 * half;
      const unsigned char mb0 = mask[(size_t)m * NKR + kb + lm];
      const unsigned char mb1 = mask[(size_t)m * NKR + kb + 16 + lm];
      if (mb0) s0[r] = NEGV;
      if (mb1) s1[r] = NEGV;
    }

    // ---- online softmax: a row spans one 16-lane half per C VGPR
    float alpha[8];
#pragma unroll
    for (int r = 0; r < 8; ++r) {
      float mx = fmaxf(s0[r], s1[r]);
      mx = fmaxf(mx, __shfl_xor(mx, 1, 32));
      mx = fmaxf(mx, __shfl_xor(mx, 2, 32));
      mx = fmaxf(mx, __shfl_xor(mx, 4, 32));
      mx = fmaxf(mx, __shfl_xor(mx, 8, 32));
      const float mn = fmaxf(mrow[r], mx);
      alpha[r] = __expf(mrow[r] - mn);           // first iter: exp(-inf)=0
      mrow[r] = mn;
      s0[r] = __expf(s0[r] - mn);
      s1[r] = __expf(s1[r] - mn);
      float rs = s0[r] + s1[r];
      rs += __shfl_xor(rs, 1, 32);
      rs += __shfl_xor(rs, 2, 32);
      rs += __shfl_xor(rs, 4, 32);
      rs += __shfl_xor(rs, 8, 32);
      lrow[r] = alpha[r] * lrow[r] + rs;
    }

    // ---- P C-layout -> A-layout via per-wave LDS patch (same-wave DS in-order)
    unsigned short* pw = Ps + wave * (16 * 32);
#pragma unroll
    for (int r = 0; r < 8; ++r) {
      const int m = r + 8 * half;
      pw[m * 32 + lm]      = f2bfu(s0[r]);
      pw[m * 32 + 16 + lm] = f2bfu(s1[r]);
    }
    asm volatile("s_wait_dscnt 0" ::: "memory");
    union { bf16x16 v; uint4 u[2]; } PA;
    {
      const unsigned short* prow = pw + lm * 32;
      PA.u[0] = *(const uint4*)(prow + (half ? 8 : 0));
      PA.u[1] = *(const uint4*)(prow + (half ? 24 : 16));
    }

    // ---- O = alpha*O + P(16x32) @ V(32x512): 32 independent C-tile chains
#pragma unroll
    for (int t = 0; t < 32; ++t) {
#pragma unroll
      for (int r = 0; r < 8; ++r) o[t][r] *= alpha[r];
      union { bf16x16 v; unsigned u[8]; } BV;
      const unsigned short* vrow = Vs + t * 16 + lm;    // n = lm, kk in table order
#pragma unroll
      for (int vv = 0; vv < 8; ++vv) {
        const int kk = (vv < 4) ? (8 * half + 2 * vv) : (16 + 8 * half + 2 * (vv - 4));
        BV.u[vv] = (unsigned)vrow[kk * DDIM] | ((unsigned)vrow[(kk + 1) * DDIM] << 16);
      }
      o[t] = __builtin_amdgcn_wmma_f32_16x16x32_bf16(false, PA.v, false, BV.v,
                                                     (short)0, o[t], false, false);
    }
    __syncthreads();                             // protect Kt/V buffers
  }

  // ---- epilogue: normalize and store f32
#pragma unroll
  for (int r = 0; r < 8; ++r) {
    const float inv = 1.0f / lrow[r];
    const int m = q0 + r + 8 * half;
#pragma unroll
    for (int t = 0; t < 32; ++t)
      Out[(size_t)m * DDIM + t * 16 + lm] = o[t][r] * inv;
  }
}

// ---------------------------------------------------------------------------
extern "C" void kernel_launch(void* const* d_in, const int* in_sizes, int n_in,
                              void* d_out, int out_size, void* d_ws, size_t ws_size,
                              hipStream_t stream) {
  (void)in_sizes; (void)n_in; (void)out_size; (void)ws_size;
  const float* Q   = (const float*)d_in[0];
  const float* K   = (const float*)d_in[1];
  const float* V   = (const float*)d_in[2];
  const float* WQw = (const float*)d_in[3];
  const float* WQb = (const float*)d_in[4];
  const float* WKw = (const float*)d_in[5];
  const float* WKb = (const float*)d_in[6];
  const unsigned char* mask = (const unsigned char*)d_in[7];
  float* out = (float*)d_out;

  unsigned short* Qp  = (unsigned short*)d_ws;           // 8192*512 bf16
  unsigned short* Kp  = Qp + (size_t)NQR * DDIM;         // 8192*512 bf16
  unsigned short* Vbf = Kp + (size_t)NKR * DDIM;         // 8192*512 bf16

  dim3 pb(256);
  dim3 pg(NQR / 16, DDIM / 128);                         // 512 x 4
  proj16<<<pg, pb, 0, stream>>>(Q, WQw, WQb, Qp);
  proj16<<<pg, pb, 0, stream>>>(K, WKw, WKb, Kp);
  cvt_bf16<<<(NKR * DDIM) / (4 * 256), 256, 0, stream>>>(V, Vbf);
  flash_attn<<<NQR / 128, 256, SM_SZ, stream>>>(Qp, Kp, Vbf, mask, out);
}
